// Set2Set_4157528342960
// MI455X (gfx1250) — matrix operand order repaired and checked by reading, not compiled, not run on hardware
//
#include <hip/hip_runtime.h>

#define B_     128
#define N_     1024
#define D_     256
#define TWO_D  512
#define FOUR_D 1024
#define T_ITERS 10

typedef __attribute__((ext_vector_type(16))) _Float16 v16h;
typedef __attribute__((ext_vector_type(8)))  float    v8f;

// ---------------------------------------------------------------------------
// Workspace layout (float offsets):
//   h_s    @ 0        (32768)
//   c_s    @ 32768    (32768)
//   q_s    @ 65536    (65536)
//   gates  @ 131072   (131072)
//   e_s    @ 262144   (131072)   scores -> normalized attention (in place)
//   r_part @ 393216   (262144)   8 partial readouts per (b,d)
//   Wih16  @ 655360   (524288 halves)
//   Whh16  @ 917504   (262144 halves)
//   ---- 4 MiB ----
//   x16    @ 4 MiB    (33554432 halves = 64 MiB)   [optional, ws permitting]
// ---------------------------------------------------------------------------

__global__ void zero_state(float* __restrict__ p) {
    int i = blockIdx.x * blockDim.x + threadIdx.x;
    p[i] = 0.0f;                          // zeroes h, c, q
}

__global__ void cvt_weights(const float* __restrict__ Wih,
                            const float* __restrict__ Whh,
                            _Float16* __restrict__ Wih16,
                            _Float16* __restrict__ Whh16)
{
    int i = blockIdx.x * blockDim.x + threadIdx.x;     // up to 524288
    Wih16[i] = (_Float16)Wih[i];
    if (i < FOUR_D * D_) Whh16[i] = (_Float16)Whh[i];
}

__global__ void cvt_x(const float* __restrict__ x, _Float16* __restrict__ x16) {
    size_t i = (size_t)(blockIdx.x * blockDim.x + threadIdx.x) * 4;
    float4 v = *(const float4*)(x + i);
    x16[i + 0] = (_Float16)v.x;
    x16[i + 1] = (_Float16)v.y;
    x16[i + 2] = (_Float16)v.z;
    x16[i + 3] = (_Float16)v.w;
}

// ---------------------------------------------------------------------------
// gates[B,4D] = q[B,2D] @ W_ih^T + h[B,D] @ W_hh^T   (f16 in, f32 accumulate)
// One wave per 16x16 tile. Fragment layouts per CDNA5 ISA 7.12.2:
//   A: lane l -> M=l&15, half=l>>4; elem e: k = (e<8)? half*8+e : 16+half*8+e-8
//   B: lane l -> N=l&15, half=l>>4; elem e: k = (e<8)? half*8+e : 16+half*8+e-8
//   D: lane l -> N=l&15; vgpr r -> M=(l>>4)*8+r
// ---------------------------------------------------------------------------
__global__ void lstm_gates_wmma(const float* __restrict__ q,
                                const float* __restrict__ h,
                                const _Float16* __restrict__ Wih16, // [4D,2D]
                                const _Float16* __restrict__ Whh16, // [4D,D]
                                float* __restrict__ gates)          // [B,4D]
{
    const int wave = threadIdx.x >> 5;
    const int lane = threadIdx.x & 31;
    const int tile = blockIdx.x * 8 + wave;   // 512 tiles
    const int tM   = tile & 7;
    const int tN   = tile >> 3;
    const int l16  = lane & 15;
    const int half = lane >> 4;
    const int m0   = tM * 16;
    const int n0   = tN * 16;

    v8f acc = {};

    {   // q @ W_ih^T : K = 512
        const float*    qrow = q     + (size_t)(m0 + l16) * TWO_D;
        const _Float16* wrow = Wih16 + (size_t)(n0 + l16) * TWO_D;
        for (int kb = 0; kb < TWO_D; kb += 32) {
            v16h a, bm;
#pragma unroll
            for (int e = 0; e < 16; ++e) {
                int ka = kb + ((e < 8) ? (half * 8 + e) : (16 + half * 8 + (e - 8)));
                a[e]  = (_Float16)qrow[ka];
                bm[e] = wrow[ka];
            }
            acc = __builtin_amdgcn_wmma_f32_16x16x32_f16(
                      false, a, false, bm, (short)0, acc, false, false);
        }
    }
    {   // h @ W_hh^T : K = 256
        const float*    hrow = h     + (size_t)(m0 + l16) * D_;
        const _Float16* wrow = Whh16 + (size_t)(n0 + l16) * D_;
        for (int kb = 0; kb < D_; kb += 32) {
            v16h a, bm;
#pragma unroll
            for (int e = 0; e < 16; ++e) {
                int ka = kb + ((e < 8) ? (half * 8 + e) : (16 + half * 8 + (e - 8)));
                a[e]  = (_Float16)hrow[ka];
                bm[e] = wrow[ka];
            }
            acc = __builtin_amdgcn_wmma_f32_16x16x32_f16(
                      false, a, false, bm, (short)0, acc, false, false);
        }
    }
#pragma unroll
    for (int r = 0; r < 8; ++r) {
        int m = m0 + half * 8 + r;
        gates[(size_t)m * FOUR_D + n0 + l16] = acc[r];
    }
}

// ---------------------------------------------------------------------------
__device__ __forceinline__ float sigmoidf_(float v) {
    return 1.0f / (1.0f + __expf(-v));
}

__global__ void lstm_pointwise(const float* __restrict__ gates,
                               const float* __restrict__ b_ih,
                               const float* __restrict__ b_hh,
                               float* __restrict__ h,
                               float* __restrict__ c,
                               float* __restrict__ q)
{
    int idx = blockIdx.x * blockDim.x + threadIdx.x;   // [0, B*D)
    int b = idx >> 8;
    int d = idx & 255;
    const float* g = gates + (size_t)b * FOUR_D;
    float gi = g[d]        + b_ih[d]        + b_hh[d];
    float gf = g[D_ + d]   + b_ih[D_ + d]   + b_hh[D_ + d];
    float gg = g[2*D_ + d] + b_ih[2*D_ + d] + b_hh[2*D_ + d];
    float go = g[3*D_ + d] + b_ih[3*D_ + d] + b_hh[3*D_ + d];
    float i = sigmoidf_(gi), f = sigmoidf_(gf), o = sigmoidf_(go);
    float gt = tanhf(gg);
    float cn = f * c[idx] + i * gt;
    float hn = o * tanhf(cn);
    c[idx] = cn;
    h[idx] = hn;
    q[(size_t)b * TWO_D + d] = hn;
}

// ---------------------------------------------------------------------------
// e[b,n] = mask ? -1e9 : dot(x[b,n,:], h[b,:]).  One wave per 8 rows.
// Lane owns d = lane*8..lane*8+7 -> one b128 (f16) / two b128 (f32) per row,
// 512B contiguous per wave; 5-step wave32 shfl_xor reduction.
// ---------------------------------------------------------------------------
template <typename XT>
__global__ void attn_scores_t(const XT* __restrict__ x,
                              const float* __restrict__ h,
                              const int*  __restrict__ mask,
                              float* __restrict__ e)
{
    const int b     = blockIdx.y;
    const int chunk = blockIdx.x;            // 16 chunks of 64 rows
    const int wave  = threadIdx.x >> 5;
    const int lane  = threadIdx.x & 31;

    float hreg[8];
#pragma unroll
    for (int j = 0; j < 8; ++j)
        hreg[j] = h[(size_t)b * D_ + lane * 8 + j];

    const size_t xb = (size_t)b * N_ * D_;
#pragma unroll
    for (int i = 0; i < 8; ++i) {
        int n = chunk * 64 + wave * 8 + i;
        const XT* xr = x + xb + (size_t)n * D_ + lane * 8;
        float s = 0.0f;
#pragma unroll
        for (int j = 0; j < 8; ++j)
            s = fmaf((float)xr[j], hreg[j], s);
#pragma unroll
        for (int off = 16; off > 0; off >>= 1)
            s += __shfl_xor(s, off, 32);
        if (lane == 0) {
            size_t p = (size_t)b * N_ + n;
            e[p] = mask[p] ? -1.0e9f : s;
        }
    }
}

// ---------------------------------------------------------------------------
// In-place masked softmax over each row of e[B,N]. 256 threads per row.
// ---------------------------------------------------------------------------
__global__ void softmax_norm(float* __restrict__ e)
{
    __shared__ float red[8];
    const int b    = blockIdx.x;
    const int tid  = threadIdx.x;
    const int lane = tid & 31, wave = tid >> 5;

    float ev[4];
    float m = -3.0e38f;
#pragma unroll
    for (int i = 0; i < 4; ++i) {
        ev[i] = e[(size_t)b * N_ + tid + 256 * i];
        m = fmaxf(m, ev[i]);
    }
#pragma unroll
    for (int off = 16; off > 0; off >>= 1)
        m = fmaxf(m, __shfl_xor(m, off, 32));
    if (lane == 0) red[wave] = m;
    __syncthreads();
    float mx = red[0];
#pragma unroll
    for (int w = 1; w < 8; ++w) mx = fmaxf(mx, red[w]);

    float s = 0.0f;
#pragma unroll
    for (int i = 0; i < 4; ++i) {
        ev[i] = __expf(ev[i] - mx);
        s += ev[i];
    }
#pragma unroll
    for (int off = 16; off > 0; off >>= 1)
        s += __shfl_xor(s, off, 32);
    __syncthreads();
    if (lane == 0) red[wave] = s;
    __syncthreads();
    float sum = 0.0f;
#pragma unroll
    for (int w = 0; w < 8; ++w) sum += red[w];
    float inv = 1.0f / sum;
#pragma unroll
    for (int i = 0; i < 4; ++i)
        e[(size_t)b * N_ + tid + 256 * i] = ev[i] * inv;
}

// ---------------------------------------------------------------------------
// Readout split over 8 N-chunks for occupancy (1024 workgroups), then a
// deterministic tree reduction (no float atomics -> bitwise reproducible).
// ---------------------------------------------------------------------------
template <typename XT>
__global__ void readout_partial_t(const float* __restrict__ a,   // [B,N] normalized
                                  const XT* __restrict__ x,
                                  float* __restrict__ r_part)    // [B,8,D]
{
    const int chunk = blockIdx.x;   // 0..7
    const int b     = blockIdx.y;   // 0..127
    const int d     = threadIdx.x;  // 0..255
    __shared__ float a_sh[128];
    if (d < 128) a_sh[d] = a[(size_t)b * N_ + chunk * 128 + d];
    __syncthreads();

    const XT* xb = x + (size_t)b * N_ * D_ + (size_t)(chunk * 128) * D_ + d;
    float r = 0.0f;
    for (int n = 0; n < 128; ++n)
        r = fmaf(a_sh[n], (float)xb[(size_t)n * D_], r);
    r_part[((size_t)b * 8 + chunk) * D_ + d] = r;
}

__global__ void readout_reduce(const float* __restrict__ r_part,
                               float* __restrict__ q)
{
    int idx = blockIdx.x * blockDim.x + threadIdx.x;   // [0, B*D)
    int b = idx >> 8;
    int d = idx & 255;
    float r = 0.0f;
#pragma unroll
    for (int cc = 0; cc < 8; ++cc)
        r += r_part[((size_t)b * 8 + cc) * D_ + d];
    q[(size_t)b * TWO_D + D_ + d] = r;
}

__global__ void copy_out(const float* __restrict__ src, float* __restrict__ dst) {
    int i = blockIdx.x * blockDim.x + threadIdx.x;
    dst[i] = src[i];
}

// ---------------------------------------------------------------------------
extern "C" void kernel_launch(void* const* d_in, const int* in_sizes, int n_in,
                              void* d_out, int out_size, void* d_ws, size_t ws_size,
                              hipStream_t stream)
{
    (void)in_sizes; (void)n_in; (void)out_size;
    const float* x    = (const float*)d_in[0];
    const int*   mask = (const int*)  d_in[1];
    const float* Wih  = (const float*)d_in[2];
    const float* Whh  = (const float*)d_in[3];
    const float* bih  = (const float*)d_in[4];
    const float* bhh  = (const float*)d_in[5];

    float* ws     = (float*)d_ws;
    float* h_s    = ws;
    float* c_s    = ws + 32768;
    float* q_s    = ws + 65536;
    float* gates  = ws + 131072;
    float* e_s    = ws + 262144;
    float* r_part = ws + 393216;
    _Float16* Wih16 = (_Float16*)(ws + 655360);
    _Float16* Whh16 = (_Float16*)(ws + 917504);
    _Float16* x16   = (_Float16*)((char*)d_ws + (4ull << 20));

    // f16 copy of x (halves dominant L2 traffic) only if scratch allows.
    const bool use16 = ws_size >= (4ull << 20) + (64ull << 20);

    zero_state<<<512, 256, 0, stream>>>(ws);                 // h, c, q = 0
    cvt_weights<<<2048, 256, 0, stream>>>(Wih, Whh, Wih16, Whh16);
    if (use16)
        cvt_x<<<32768, 256, 0, stream>>>(x, x16);            // 33.5M elems, x4/thread

    for (int t = 0; t < T_ITERS; ++t) {
        lstm_gates_wmma<<<64, 256, 0, stream>>>(q_s, h_s, Wih16, Whh16, gates);
        lstm_pointwise<<<128, 256, 0, stream>>>(gates, bih, bhh, h_s, c_s, q_s);
        if (use16) {
            attn_scores_t<_Float16><<<dim3(16, B_), 256, 0, stream>>>(x16, h_s, mask, e_s);
            softmax_norm<<<B_, 256, 0, stream>>>(e_s);
            readout_partial_t<_Float16><<<dim3(8, B_), 256, 0, stream>>>(e_s, x16, r_part);
        } else {
            attn_scores_t<float><<<dim3(16, B_), 256, 0, stream>>>(x, h_s, mask, e_s);
            softmax_norm<<<B_, 256, 0, stream>>>(e_s);
            readout_partial_t<float><<<dim3(8, B_), 256, 0, stream>>>(e_s, x, r_part);
        }
        readout_reduce<<<128, 256, 0, stream>>>(r_part, q_s);
    }
    copy_out<<<256, 256, 0, stream>>>(q_s, (float*)d_out);
}
